// SeqlenDynamicSparseAttention_54743653155168
// MI455X (gfx1250) — matrix operand order repaired and checked by reading, not compiled, not run on hardware
//
#include <hip/hip_runtime.h>

typedef __attribute__((ext_vector_type(16))) _Float16 v16h;
typedef __attribute__((ext_vector_type(8)))  _Float16 v8h;
typedef __attribute__((ext_vector_type(8)))  float    v8f;

#define NEGMASK (-1e9f)
#define LDK 72   // padded LDS row stride in halfs (144B: 16B aligned, bank-spread)

__device__ __forceinline__ v16h ld16(const _Float16* p){
  v8h a = *(const v8h*)p;
  v8h b = *(const v8h*)(p + 8);
  v16h r;
  #pragma unroll
  for(int i=0;i<8;++i){ r[i]=a[i]; r[i+8]=b[i]; }
  return r;
}

__device__ __forceinline__ float rmax16(float v){
  #pragma unroll
  for(int m=8;m>=1;m>>=1) v = fmaxf(v, __shfl_xor(v, m, 32));
  return v;
}
__device__ __forceinline__ float rsum16(float v){
  #pragma unroll
  for(int m=8;m>=1;m>>=1) v += __shfl_xor(v, m, 32);
  return v;
}

__device__ __forceinline__ v8h cvt8(float4 a, float4 b){
  v8h r;
  r[0]=(_Float16)a.x; r[1]=(_Float16)a.y; r[2]=(_Float16)a.z; r[3]=(_Float16)a.w;
  r[4]=(_Float16)b.x; r[5]=(_Float16)b.y; r[6]=(_Float16)b.z; r[7]=(_Float16)b.w;
  return r;
}

__launch_bounds__(256)
__global__ void seqlen_attn_wmma_kernel(const float* __restrict__ Q,
                                        const float* __restrict__ K,
                                        const float* __restrict__ V,
                                        const int*   __restrict__ seqlens,
                                        float*       __restrict__ Out,
                                        int H, int N, int E)
{
  // E == 64 assumed (reference shape)
  __shared__ __align__(16) _Float16 Ksh [64*LDK];      // [key][e]
  __shared__ __align__(16) _Float16 VshT[64*LDK];      // [e][key] (transposed)
  __shared__ __align__(16) _Float16 Psh [8*16*LDK];    // per-wave P scratch [wave*16+row][key]

  const int qblk = blockIdx.x;
  const int h    = blockIdx.y;
  const int b    = blockIdx.z;
  const int tid  = threadIdx.x;
  const int w    = tid >> 5;
  const int lane = tid & 31;
  const int ln   = lane & 15;
  const int hi   = lane >> 4;   // 0: lanes 0-15, 1: lanes 16-31

  const int sl = seqlens[b];
  const int q0 = qblk*128 + w*16;

  const size_t bh = ((size_t)b*H + h) * (size_t)N * (size_t)E;

  // ---- Q A-fragments (rows q0..q0+15, e=0..63), f16 A layout (16x32 per frag) ----
  const float* qp = Q + bh + (size_t)(q0 + ln)*E;
  v16h aq[2];
  #pragma unroll
  for(int f=0; f<2; ++f){
    #pragma unroll
    for(int t=0; t<16; ++t){
      int col = 32*f + ((t<8) ? t : t+8) + 8*hi;   // ISA 16-bit A-matrix 16x32 map
      aq[f][t] = (_Float16)qp[col];
    }
  }

  v8f o[4];
  #pragma unroll
  for(int n=0;n<4;++n) o[n] = (v8f){0,0,0,0,0,0,0,0};
  float mrow[8], lrow[8];
  #pragma unroll
  for(int r=0;r<8;++r){ mrow[r] = -1e30f; lrow[r] = 0.f; }

  const int kb_end = (sl + 63) >> 6;   // only tiles containing >=1 valid key

  // per-thread tile-loader coordinates
  const int lr  = tid >> 2;            // row 0..63
  const int lc0 = (tid & 3) * 16;      // col chunk 0/16/32/48

  for(int j=0; j<kb_end; ++j){
    const int key0 = j*64;
    __syncthreads();   // protect LDS tiles from previous iteration's readers
    {
      // cooperative 64x64 fp32 tile loads (b128), f16 convert; V stored transposed
      const float4* kp4 = (const float4*)(K + bh + (size_t)(key0 + lr)*E + lc0);
      const float4* vp4 = (const float4*)(V + bh + (size_t)(key0 + lr)*E + lc0);
      float4 k0 = kp4[0], k1 = kp4[1], k2 = kp4[2], k3 = kp4[3];
      float4 v0 = vp4[0], v1 = vp4[1], v2 = vp4[2], v3 = vp4[3];

      *(v8h*)&Ksh[lr*LDK + lc0]     = cvt8(k0, k1);   // ds_store_b128
      *(v8h*)&Ksh[lr*LDK + lc0 + 8] = cvt8(k2, k3);   // ds_store_b128

      float vv[16] = {v0.x,v0.y,v0.z,v0.w, v1.x,v1.y,v1.z,v1.w,
                      v2.x,v2.y,v2.z,v2.w, v3.x,v3.y,v3.z,v3.w};
      #pragma unroll
      for(int i=0;i<16;++i)
        VshT[(lc0+i)*LDK + lr] = (_Float16)vv[i];     // transposed scatter (b16)
    }
    // prefetch next key tile into L2/L0 while this tile is consumed
    if(j + 1 < kb_end){
      const char* kn = (const char*)(K + bh + (size_t)(key0 + 64 + lr)*E + lc0);
      const char* vn = (const char*)(V + bh + (size_t)(key0 + 64 + lr)*E + lc0);
      __builtin_prefetch(kn, 0, 1);                   // global_prefetch_b8
      __builtin_prefetch(vn, 0, 1);
    }
    __syncthreads();

    // ---- S = Q @ K^T : four 16x16 sub-tiles (16 keys each) ----
    v8f sc[4];
    #pragma unroll
    for(int s=0;s<4;++s){
      v8f c = (v8f){0,0,0,0,0,0,0,0};
      #pragma unroll
      for(int f=0; f<2; ++f){
        // B-frag of K^T: column n = key, rows k = e (table order: e 32f+8hi..+7, 32f+16+8hi..+7)
        const _Float16* kr = &Ksh[(s*16 + ln)*LDK + 32*f + 8*hi];
        v8h kb0 = *(const v8h*)kr; v8h kb1 = *(const v8h*)(kr + 16);
        v16h bk;
        for(int i=0;i<8;++i){ bk[i]=kb0[i]; bk[i+8]=kb1[i]; }
        c = __builtin_amdgcn_wmma_f32_16x16x32_f16(false, aq[f], false, bk,
                                                   (short)0, c, false, false);
      }
      const int key = key0 + s*16 + ln;   // C-layout: all 8 VGPRs share column = key
      if(key >= sl){
        #pragma unroll
        for(int r=0;r<8;++r) c[r] = NEGMASK;
      }
      sc[s] = c;
    }

    // ---- online softmax (rows live along 16-lane halves) ----
    float scal[8];
    #pragma unroll
    for(int r=0;r<8;++r){
      float v = fmaxf(fmaxf(sc[0][r], sc[1][r]), fmaxf(sc[2][r], sc[3][r]));
      v = rmax16(v);
      const float mn  = fmaxf(mrow[r], v);
      const float scf = __expf(mrow[r] - mn);
      mrow[r] = mn;
      scal[r] = scf;
      float psum = 0.f;
      #pragma unroll
      for(int s=0;s<4;++s){
        float p = __expf(sc[s][r] - mn);
        sc[s][r] = p;
        psum += p;
      }
      lrow[r] = lrow[r]*scf + rsum16(psum);
    }
    #pragma unroll
    for(int n=0;n<4;++n){
      #pragma unroll
      for(int r=0;r<8;++r) o[n][r] *= scal[r];
    }

    // ---- C-layout -> A-layout for P via per-wave LDS scratch ----
    #pragma unroll
    for(int s=0;s<4;++s){
      #pragma unroll
      for(int r=0;r<8;++r){
        Psh[(w*16 + r + 8*hi)*LDK + s*16 + ln] = (_Float16)sc[s][r];
      }
    }
    // same-wave LDS ops are in-order; compiler inserts dscnt waits for the RAW dep

    const _Float16* pp = &Psh[(w*16 + ln)*LDK];
    v16h pa[2];
    #pragma unroll
    for(int f=0; f<2; ++f){
      v8h c0 = *(const v8h*)(pp + 32*f + 8*hi);        // keys 32f+8hi .. +7
      v8h c1 = *(const v8h*)(pp + 32*f + 16 + 8*hi);   // keys 32f+16+8hi .. +7
      #pragma unroll
      for(int i=0;i<8;++i){ pa[f][i]=c0[i]; pa[f][i+8]=c1[i]; }
    }

    // ---- O += P @ V : four 16-column output groups ----
    #pragma unroll
    for(int n=0;n<4;++n){
      #pragma unroll
      for(int f=0; f<2; ++f){
        // B-frag of V: column n = e, rows k = key (table order via transposed store)
        const _Float16* vr = &VshT[(n*16 + ln)*LDK + 32*f + 8*hi];
        v8h vb0 = *(const v8h*)vr; v8h vb1 = *(const v8h*)(vr + 16);
        v16h vb;
        for(int i=0;i<8;++i){ vb[i]=vb0[i]; vb[i+8]=vb1[i]; }
        o[n] = __builtin_amdgcn_wmma_f32_16x16x32_f16(false, pa[f], false, vb,
                                                      (short)0, o[n], false, false);
      }
    }
  }

  // ---- epilogue: normalize, zero invalid query rows, store fp32 ----
  #pragma unroll
  for(int r=0;r<8;++r){
    const int   qrow  = q0 + r + 8*hi;
    const float inv   = (lrow[r] > 0.f) ? (1.f / lrow[r]) : 0.f;
    const bool  valid = qrow < sl;
    #pragma unroll
    for(int n=0;n<4;++n){
      Out[bh + (size_t)qrow*E + n*16 + ln] = valid ? (o[n][r] * inv) : 0.f;
    }
  }
}

extern "C" void kernel_launch(void* const* d_in, const int* in_sizes, int n_in,
                              void* d_out, int out_size, void* d_ws, size_t ws_size,
                              hipStream_t stream) {
  const float* Q  = (const float*)d_in[0];
  const float* K  = (const float*)d_in[1];
  const float* V  = (const float*)d_in[2];
  const int* seql = (const int*)d_in[3];
  float* out      = (float*)d_out;

  const int B = in_sizes[3];          // seqlens has B entries
  const int H = 8;                    // reference shape
  const int E = 64;                   // reference shape
  const long total = (long)in_sizes[0];
  const int N = (int)(total / ((long)B * H * E));   // 2048

  dim3 grid(N / 128, H, B);
  seqlen_attn_wmma_kernel<<<grid, 256, 0, stream>>>(Q, K, V, seql, out, H, N, E);
}
